// MoleculeEncoder_16595753632016
// MI455X (gfx1250) — matrix operand order repaired and checked by reading, not compiled, not run on hardware
//
#include <hip/hip_runtime.h>
#include <hip/hip_bf16.h>
#include <math.h>

// MI455X (gfx1250) implementation.
// Strategy (memory-bound workload, 23.3 TB/s HBM, WMMA-only matrix ops):
//  - all GEMMs via V_WMMA_F32_16X16X32_F16 (f16 operands, f32 accum)
//  - operands pre-packed into the exact WMMA per-lane VGPR layouts so each
//    wave issues plain contiguous 32B/lane loads (no in-kernel shuffling)
//  - concat inputs (GNN message [x[row]||ea], spatial scalars||bond,
//    [mol_feats||seq]) are fused into the packing kernels (never materialized)
//  - A tiles moved memory->LDS with GLOBAL_LOAD_ASYNC_TO_LDS_B64 (ASYNCcnt),
//    double-buffered so the copy of tile kt+1 overlaps the WMMA of tile kt,
//    one workgroup barrier per K-step
//  - LN/GELU/residual epilogues: one wave per row, wave32 shfl reductions

typedef __attribute__((ext_vector_type(16))) _Float16 v16h;
typedef __attribute__((ext_vector_type(8)))  float    v8f;

// K index held by (lane, half) for the 16-bit 16x32 A layout (ISA 05_wmma 7.12.2)
__device__ __forceinline__ int kmapA(int lane, int h) {
  return (h < 8 ? h : h + 8) + ((lane >= 16) ? 8 : 0);
}

__device__ __forceinline__ float gelu_exact(float t) {
  return 0.5f * t * (1.0f + erff(t * 0.70710678118654752f));
}

// gfx1250 async memory->LDS copy (8B per lane), tracked by ASYNCcnt.
__device__ __forceinline__ void async_copy_b64(unsigned lds_off, unsigned long long gaddr) {
  asm volatile("global_load_async_to_lds_b64 %0, %1, off"
               :: "v"(lds_off), "v"(gaddr)
               : "memory");
}
__device__ __forceinline__ void wait_asynccnt0() {
  asm volatile("s_wait_asynccnt 0x0" ::: "memory");
}

// ---------------------------------------------------------------------------
// Packing kernels: produce f16 tiles laid out so that lane L of a wave loads
// its 16 halves (32B) contiguously and they land exactly in the WMMA operand
// registers.
//   A tile (16x32, MxK): out[((mt*Kt+kt)*32+lane)*16+h] = A[mt*16+lane%16, kt*32+kmapA]
//   B tile (32x16, KxN): out[((kt*Nt+nt)*32+lane)*16+h] = W[nt*16+lane%16, kt*32+kmapA]
// ---------------------------------------------------------------------------
__global__ void pack_w_kernel(const float* __restrict__ W, _Float16* __restrict__ out,
                              int O, int I, int Kpad) {
  size_t t = (size_t)blockIdx.x * blockDim.x + threadIdx.x;
  size_t total = (size_t)(Kpad >> 5) * (size_t)(O >> 4) * 512;
  if (t >= total) return;
  int h = (int)(t & 15), lane = (int)((t >> 4) & 31);
  size_t tile = t >> 9;
  int Ntl = O >> 4;
  int nt = (int)(tile % Ntl), kt = (int)(tile / Ntl);
  int n = nt * 16 + (lane & 15);
  int k = kt * 32 + kmapA(lane, h);
  float v = (k < I) ? W[(size_t)n * I + k] : 0.f;
  out[t] = (_Float16)v;
}

__global__ void pack_a_kernel(const float* __restrict__ A, _Float16* __restrict__ out,
                              int Mtiles, int K, int Kpad, int ldA, int col0) {
  size_t t = (size_t)blockIdx.x * blockDim.x + threadIdx.x;
  int Kt = Kpad >> 5;
  size_t total = (size_t)Mtiles * Kt * 512;
  if (t >= total) return;
  int h = (int)(t & 15), lane = (int)((t >> 4) & 31);
  size_t tile = t >> 9;
  int kt = (int)(tile % Kt), mt = (int)(tile / Kt);
  int m = mt * 16 + (lane & 15);
  int k = kt * 32 + kmapA(lane, h);
  float v = (k < K) ? A[(size_t)m * ldA + col0 + k] : 0.f;
  out[t] = (_Float16)v;
}

// GNN message input: [ x[row[e]] (512) || ea[e] (512) ]  -> K = 1024, Kt = 32
__global__ void pack_a_gnn_kernel(const float* __restrict__ x, const float* __restrict__ ea,
                                  const int* __restrict__ rowidx, _Float16* __restrict__ out,
                                  int Mtiles) {
  size_t t = (size_t)blockIdx.x * blockDim.x + threadIdx.x;
  size_t total = (size_t)Mtiles * 32 * 512;
  if (t >= total) return;
  int h = (int)(t & 15), lane = (int)((t >> 4) & 31);
  size_t tile = t >> 9;
  int kt = (int)(tile & 31), mt = (int)(tile >> 5);
  int e = mt * 16 + (lane & 15);
  int k = kt * 32 + kmapA(lane, h);
  float v = (k < 512) ? x[(size_t)rowidx[e] * 512 + k]
                      : ea[(size_t)e * 512 + (k - 512)];
  out[t] = (_Float16)v;
}

// spatial input: [dist, theta, phi, tau, bond(128), pad..160] -> Kt = 5
__global__ void pack_a_spatial_kernel(const float* __restrict__ dist, const float* __restrict__ theta,
                                      const float* __restrict__ phi, const float* __restrict__ tau,
                                      const float* __restrict__ bond, _Float16* __restrict__ out,
                                      int Mtiles) {
  size_t t = (size_t)blockIdx.x * blockDim.x + threadIdx.x;
  size_t total = (size_t)Mtiles * 5 * 512;
  if (t >= total) return;
  int h = (int)(t & 15), lane = (int)((t >> 4) & 31);
  size_t tile = t >> 9;
  int kt = (int)(tile % 5), mt = (int)(tile / 5);
  int e = mt * 16 + (lane & 15);
  int k = kt * 32 + kmapA(lane, h);
  float v = 0.f;
  if (k == 0) v = dist[e];
  else if (k == 1) v = theta[e];
  else if (k == 2) v = phi[e];
  else if (k == 3) v = tau[e];
  else if (k < 132) v = bond[(size_t)e * 128 + (k - 4)];
  out[t] = (_Float16)v;
}

// fusion input: [ mol_feats[i] (512) || seq (512, broadcast) ] -> Kt = 32
__global__ void pack_a_combined_kernel(const float* __restrict__ xf, const float* __restrict__ seq,
                                       _Float16* __restrict__ out, int Mtiles) {
  size_t t = (size_t)blockIdx.x * blockDim.x + threadIdx.x;
  size_t total = (size_t)Mtiles * 32 * 512;
  if (t >= total) return;
  int h = (int)(t & 15), lane = (int)((t >> 4) & 31);
  size_t tile = t >> 9;
  int kt = (int)(tile & 31), mt = (int)(tile >> 5);
  int m = mt * 16 + (lane & 15);
  int k = kt * 32 + kmapA(lane, h);
  float v = (k < 512) ? xf[(size_t)m * 512 + k] : seq[k - 512];
  out[t] = (_Float16)v;
}

// ---------------------------------------------------------------------------
// WMMA GEMM: C[M,N] = A[M,K] * B[K,N] (operands packed f16, f32 accum).
// Block = 128 threads = 4 waves; block covers one 16-row strip x 4 N-tiles.
// A tile is DMA'd memory->LDS (async, double-buffered): each lane copies 8B,
// each wave 256B, 4 waves = one 1KB 16x32 f16 tile. The copy of tile kt+1 is
// issued right after the barrier and overlaps the B loads + WMMA of tile kt.
// Requires: Ntiles % 4 == 0 (true here: N in {128,256,512,1536,2048}).
// ---------------------------------------------------------------------------
__global__ void __launch_bounds__(128) wmma_gemm_kernel(const _Float16* __restrict__ Ap,
                                                        const _Float16* __restrict__ Bp,
                                                        float* __restrict__ C,
                                                        int Ntiles, int Ktiles) {
  __shared__ alignas(32) _Float16 As[2][512];  // 2 x 1KB A tiles
  const int mt   = blockIdx.x;
  const int nt   = blockIdx.y * 4 + (threadIdx.x >> 5);
  const int lane = threadIdx.x & 31;
  const int tid  = threadIdx.x;

  // per-lane LDS byte offsets (generic-pointer low 32 bits = LDS offset)
  const unsigned lds0 = (unsigned)(uintptr_t)(&As[0][0]) + (unsigned)tid * 8u;
  const unsigned lds1 = lds0 + 1024u;
  unsigned long long gsrc =
      (unsigned long long)(uintptr_t)(Ap + (size_t)mt * Ktiles * 512) +
      (unsigned long long)tid * 8ull;

  v8f acc = {};
  const _Float16* Bt = Bp + ((size_t)nt * 32 + lane) * 16;

  async_copy_b64(lds0, gsrc);  // stage tile 0
  for (int kt = 0; kt < Ktiles; ++kt) {
    wait_asynccnt0();     // own chunk of As[kt&1] landed in LDS
    __syncthreads();      // all waves' chunks landed; all reads of As[1-(kt&1)] done
    if (kt + 1 < Ktiles)  // overlap: DMA next tile into the idle buffer
      async_copy_b64((kt & 1) ? lds0 : lds1, gsrc + (unsigned long long)(kt + 1) * 1024ull);
    __builtin_prefetch(Bt + (size_t)(kt + 1) * Ntiles * 512, 0, 1);  // global_prefetch
    v16h av = *(const v16h*)(&As[kt & 1][0] + lane * 16);            // ds_load
    v16h bv = *(const v16h*)(Bt + (size_t)kt * Ntiles * 512);        // global_load
    acc = __builtin_amdgcn_wmma_f32_16x16x32_f16(false, av, false, bv,
                                                 (short)0, acc, false, false);
  }

  // C/D layout: VGPR r -> row r (lanes 0-15) / row r+8 (lanes 16-31), col = lane%16
  const int Nd = Ntiles * 16;
  const int row0 = (lane >> 4) << 3;
  float* cp = C + ((size_t)mt * 16 + row0) * Nd + nt * 16 + (lane & 15);
#pragma unroll
  for (int r = 0; r < 8; ++r) cp[(size_t)r * Nd] = acc[r];
}

// ---------------------------------------------------------------------------
// Epilogue: one wave per row. modes:
//  0: Y = C + b
//  1: Y = gelu(LN(C + b) * g + beta)              (enc block)
//  2: Y = LN(C) * g + beta                        (gnn_norm; bias may be null)
//  3: Y = LN(X + C + b) * g + beta                (transformer post-norm)
//  4: Y = relu(C + b)
//  5: Y = X + gelu(clip(LN(nan2num(C + b))*g+beta, +-10))   (SafeResidualBlock)
// ---------------------------------------------------------------------------
__global__ void __launch_bounds__(256) epilogue_kernel(const float* __restrict__ C,
                                                       const float* __restrict__ bias,
                                                       const float* __restrict__ X,
                                                       const float* __restrict__ gw,
                                                       const float* __restrict__ bw,
                                                       float* __restrict__ Y,
                                                       int M, int Nd, int mode) {
  const int wv = threadIdx.x >> 5, lane = threadIdx.x & 31;
  const int row = blockIdx.x * 8 + wv;
  if (row >= M) return;
  const size_t base = (size_t)row * Nd;

  float sum = 0.f, sum2 = 0.f;
  const bool needLN = (mode == 1 || mode == 2 || mode == 3 || mode == 5);
  if (needLN) {
    for (int j = lane; j < Nd; j += 32) {
      float v = C[base + j];
      if (bias) v += bias[j];
      if (mode == 3) v += X[base + j];
      if (mode == 5) {
        if (v != v) v = 0.f;
        else if (isinf(v)) v = (v > 0.f) ? 1e5f : -1e5f;
      }
      sum += v; sum2 += v * v;
    }
#pragma unroll
    for (int o = 16; o > 0; o >>= 1) {
      sum  += __shfl_xor(sum,  o, 32);
      sum2 += __shfl_xor(sum2, o, 32);
    }
  }
  const float mean = sum / (float)Nd;
  const float rstd = rsqrtf(fmaxf(sum2 / (float)Nd - mean * mean, 0.f) + 1e-5f);

  for (int j = lane; j < Nd; j += 32) {
    float v = C[base + j];
    if (bias) v += bias[j];
    float out;
    if (mode == 0) {
      out = v;
    } else if (mode == 4) {
      out = fmaxf(v, 0.f);
    } else if (mode == 5) {
      if (v != v) v = 0.f;
      else if (isinf(v)) v = (v > 0.f) ? 1e5f : -1e5f;
      float tln = (v - mean) * rstd * gw[j] + bw[j];
      tln = fminf(fmaxf(tln, -10.f), 10.f);
      out = X[base + j] + gelu_exact(tln);
    } else {  // 1, 2, 3
      if (mode == 3) v += X[base + j];
      float tln = (v - mean) * rstd * gw[j] + bw[j];
      out = (mode == 1) ? gelu_exact(tln) : tln;
    }
    Y[base + j] = out;
  }
}

// ---------------------------------------------------------------------------
// scatter-mean helpers
// ---------------------------------------------------------------------------
__global__ void zero_kernel(float* __restrict__ p, size_t n) {
  for (size_t i = (size_t)blockIdx.x * blockDim.x + threadIdx.x; i < n;
       i += (size_t)gridDim.x * blockDim.x)
    p[i] = 0.f;
}

__global__ void count_kernel(const int* __restrict__ col, float* __restrict__ counts, int E) {
  int e = blockIdx.x * blockDim.x + threadIdx.x;
  if (e < E) atomicAdd(&counts[col[e]], 1.0f);
}

__global__ void scatter_add_kernel(const float* __restrict__ m, const int* __restrict__ col,
                                   float* __restrict__ accum, int E) {
  size_t t = (size_t)blockIdx.x * blockDim.x + threadIdx.x;
  size_t total = (size_t)E * 512;
  if (t >= total) return;
  int e = (int)(t >> 9), j = (int)(t & 511);
  atomicAdd(&accum[(size_t)col[e] * 512 + j], m[t]);
}

__global__ void update_x_kernel(float* __restrict__ x, const float* __restrict__ accum,
                                const float* __restrict__ counts, int N) {
  size_t t = (size_t)blockIdx.x * blockDim.x + threadIdx.x;
  size_t total = (size_t)N * 512;
  if (t >= total) return;
  x[t] += accum[t] / fmaxf(counts[t >> 9], 1.0f);
}

__global__ void seq_mean_kernel(const float* __restrict__ h, float* __restrict__ seq, int T) {
  int j = blockIdx.x * blockDim.x + threadIdx.x;
  if (j >= 512) return;
  float s = 0.f;
  for (int r = 0; r < T; ++r) s += h[(size_t)r * 512 + j];
  seq[j] = s / (float)T;
}

// ---------------------------------------------------------------------------
// Host orchestration
// ---------------------------------------------------------------------------
extern "C" void kernel_launch(void* const* d_in, const int* in_sizes, int n_in,
                              void* d_out, int out_size, void* d_ws, size_t ws_size,
                              hipStream_t stream) {
  (void)out_size; (void)ws_size;
  if (n_in < 86) return;  // expect 7 inputs + 78 param leaves (jax sorted-key flatten) + edge_index

  const int N = in_sizes[0] / 75;   // 60000 atoms (multiple of 16)
  const int E = in_sizes[1] / 8;    // 150000 edges (multiple of 16)
  const int T = in_sizes[6] / 384;  // 256 sequence rows (multiple of 16)
  const int Mn = N / 16, Me = E / 16, Mt = T / 16;

  auto P = [&](int i) { return (const float*)d_in[i]; };
  const int* eidx = (const int*)d_in[85];
  const int* rowi = eidx;
  const int* coli = eidx + E;
  float* fout = (float*)d_out;  // also used as scratch for f (fusion output)

  // --- workspace arena (bump allocator, 256B aligned) ---
  char* wsb = (char*)d_ws;
  size_t off = 0;
  auto alloc = [&](size_t bytes) -> void* {
    void* p = wsb + off;
    off = (off + bytes + 255) & ~(size_t)255;
    return p;
  };
  auto hb = [&](size_t n) { return (_Float16*)alloc(n * sizeof(_Float16)); };
  auto fb = [&](size_t n) { return (float*)alloc(n * sizeof(float)); };

  // packed weights (persistent, ~20 MB -> resident in 192 MB L2)
  _Float16* atomWp = hb((size_t)96 * 256);
  _Float16* bondWp = hb((size_t)32 * 128);
  _Float16* spatWp = hb((size_t)160 * 256);
  _Float16* nodeWp = hb((size_t)256 * 512);
  _Float16* edgeWp = hb((size_t)256 * 512);
  _Float16* l1Wp[3], *l2Wp[3];
  for (int l = 0; l < 3; ++l) { l1Wp[l] = hb((size_t)1024 * 512); l2Wp[l] = hb((size_t)512 * 512); }
  _Float16* fusWp = hb((size_t)1024 * 512);
  _Float16* resWp = hb((size_t)512 * 512);
  _Float16* seqWp = hb((size_t)384 * 512);
  _Float16* WinWp[2], *WoutWp[2], *ff1Wp[2], *ff2Wp[2];
  for (int l = 0; l < 2; ++l) {
    WinWp[l]  = hb((size_t)512 * 1536);
    WoutWp[l] = hb((size_t)512 * 512);
    ff1Wp[l]  = hb((size_t)512 * 2048);
    ff2Wp[l]  = hb((size_t)2048 * 512);
  }

  // activation arenas (reused across phases)
  float*    x      = fb((size_t)N * 512);
  float*    counts = fb((size_t)N);
  float*    accum  = fb((size_t)N * 512);   // also holds bond_y [E,128] pre-GNN
  float*    ea     = fb((size_t)E * 512);
  _Float16* Apack  = hb((size_t)E * 1024);  // largest packed-A (307 MB)
  float*    Cbuf   = fb((size_t)E * 512);   // GEMM f32 outputs
  float*    Ybuf   = fb((size_t)E * 512);   // epilogue outputs (atom_y/spatial_y pre-GNN)
  float*    hbuf = fb((size_t)T * 512);
  float*    h2   = fb((size_t)T * 512);
  float*    qkv  = fb((size_t)T * 1536);
  float*    tb   = fb((size_t)T * 2048);
  float*    seqv = fb(512);
  float* bondY = accum;                       // [E,128]
  float* atomY = Ybuf;                        // [N,256]
  float* spatY = Ybuf + (size_t)N * 256;      // [E,256]

  // --- launch helpers ---
  auto packw = [&](const float* W, _Float16* o, int O, int I, int Kpad) {
    size_t total = (size_t)Kpad * O;
    pack_w_kernel<<<(unsigned)((total + 255) / 256), 256, 0, stream>>>(W, o, O, I, Kpad);
  };
  auto packa = [&](const float* A, _Float16* o, int Mtl, int K, int Kpad, int ldA, int col0) {
    size_t total = (size_t)Mtl * (Kpad >> 5) * 512;
    pack_a_kernel<<<(unsigned)((total + 255) / 256), 256, 0, stream>>>(A, o, Mtl, K, Kpad, ldA, col0);
  };
  auto gemm = [&](const _Float16* Ap, const _Float16* Bp, float* C, int Mtl, int Ntl, int Ktl) {
    dim3 g(Mtl, Ntl / 4);
    wmma_gemm_kernel<<<g, 128, 0, stream>>>(Ap, Bp, C, Ntl, Ktl);
  };
  auto epi = [&](const float* C, const float* b, const float* X, const float* g,
                 const float* be, float* Y, int M, int Nd, int mode) {
    epilogue_kernel<<<(M + 7) / 8, 256, 0, stream>>>(C, b, X, g, be, Y, M, Nd, mode);
  };
  auto zero = [&](float* p, size_t n) { zero_kernel<<<2048, 256, 0, stream>>>(p, n); };

  // --- pack all weights (param leaves flattened in sorted-key order, base 7) ---
  packw(P(7),  atomWp, 256, 75, 96);     // atom.W
  packw(P(11), bondWp, 128, 8, 32);      // bond.W
  packw(P(57), spatWp, 256, 132, 160);   // spatial.W
  packw(P(45), nodeWp, 512, 256, 256);   // gnn_node.W
  packw(P(17), edgeWp, 512, 256, 256);   // gnn_edge.W
  for (int l = 0; l < 3; ++l) {
    packw(P(21 + l * 8),     l1Wp[l], 512, 1024, 1024);
    packw(P(21 + l * 8 + 4), l2Wp[l], 512, 512, 512);
  }
  packw(P(15), fusWp, 512, 1024, 1024);  // fus_lin.W
  packw(P(51), resWp, 512, 512, 512);    // res.W
  packw(P(55), seqWp, 512, 384, 384);    // seq_lin.W
  for (int l = 0; l < 2; ++l) {
    int b = 61 + l * 12;
    packw(P(b + 0), WinWp[l],  1536, 512, 512);   // Win
    packw(P(b + 1), WoutWp[l], 512, 512, 512);    // Wout
    packw(P(b + 6), ff1Wp[l],  2048, 512, 512);   // ff1.W
    packw(P(b + 8), ff2Wp[l],  512, 2048, 2048);  // ff2.W
  }

  // --- bond encoder: [E,8] -> [E,128]
  packa(P(1), Apack, Me, 8, 32, 8, 0);
  gemm(Apack, bondWp, Cbuf, Me, 8, 1);
  epi(Cbuf, P(12), nullptr, P(14), P(13), bondY, E, 128, 1);

  // --- spatial encoder: [E,132] -> [E,256] (fused concat packing)
  {
    size_t total = (size_t)Me * 5 * 512;
    pack_a_spatial_kernel<<<(unsigned)((total + 255) / 256), 256, 0, stream>>>(
        P(2), P(3), P(4), P(5), bondY, Apack, Me);
  }
  gemm(Apack, spatWp, Cbuf, Me, 16, 5);
  epi(Cbuf, P(58), nullptr, P(60), P(59), spatY, E, 256, 1);

  // --- atom encoder: [N,75] -> [N,256]
  packa(P(0), Apack, Mn, 75, 96, 75, 0);
  gemm(Apack, atomWp, Cbuf, Mn, 16, 3);
  epi(Cbuf, P(8), nullptr, P(10), P(9), atomY, N, 256, 1);

  // --- gnn_node: [N,256] -> x [N,512]
  packa(atomY, Apack, Mn, 256, 256, 256, 0);
  gemm(Apack, nodeWp, Cbuf, Mn, 32, 8);
  epi(Cbuf, P(46), nullptr, P(48), P(47), x, N, 512, 1);

  // --- gnn_edge: [E,256] -> ea [E,512]
  packa(spatY, Apack, Me, 256, 256, 256, 0);
  gemm(Apack, edgeWp, Cbuf, Me, 32, 8);
  epi(Cbuf, P(18), nullptr, P(20), P(19), ea, E, 512, 1);

  // --- 3 GNN message-passing layers
  for (int l = 0; l < 3; ++l) {
    int b = 21 + l * 8;  // l1.W, l1.b, l1.beta, l1.g, l2.W, l2.b, l2.beta, l2.g
    {
      size_t total = (size_t)Me * 32 * 512;
      pack_a_gnn_kernel<<<(unsigned)((total + 255) / 256), 256, 0, stream>>>(
          x, ea, rowi, Apack, Me);
    }
    gemm(Apack, l1Wp[l], Cbuf, Me, 32, 32);                       // [E,1024]x[1024,512]
    epi(Cbuf, P(b + 1), nullptr, P(b + 3), P(b + 2), Ybuf, E, 512, 1);
    packa(Ybuf, Apack, Me, 512, 512, 512, 0);
    gemm(Apack, l2Wp[l], Cbuf, Me, 32, 16);                       // [E,512]x[512,512]
    epi(Cbuf, P(b + 5), nullptr, P(b + 7), P(b + 6), Ybuf, E, 512, 1);
    // scatter_mean(m, col) added into x
    zero(accum, (size_t)N * 512);
    zero(counts, (size_t)N);
    count_kernel<<<(E + 255) / 256, 256, 0, stream>>>(coli, counts, E);
    scatter_add_kernel<<<(unsigned)(((size_t)E * 512 + 255) / 256), 256, 0, stream>>>(
        Ybuf, coli, accum, E);
    update_x_kernel<<<(unsigned)(((size_t)N * 512 + 255) / 256), 256, 0, stream>>>(
        x, accum, counts, N);
  }

  // --- gnn_norm (LN with g/beta); scatter_mean over arange(N) is identity
  epi(x, nullptr, nullptr, P(50), P(49), x, N, 512, 2);

  // --- sequence branch: Linear + 2x TransformerEncoderLayer (L=1 -> o == v)
  packa(P(6), Apack, Mt, 384, 384, 384, 0);
  gemm(Apack, seqWp, Cbuf, Mt, 32, 12);
  epi(Cbuf, P(56), nullptr, nullptr, nullptr, hbuf, T, 512, 0);
  for (int l = 0; l < 2; ++l) {
    int b = 61 + l * 12;  // Win,Wout,b1,b2,bin,bout,ff1.W,ff1.b,ff2.W,ff2.b,g1,g2
    packa(hbuf, Apack, Mt, 512, 512, 512, 0);
    gemm(Apack, WinWp[l], Cbuf, Mt, 96, 16);                      // qkv
    epi(Cbuf, P(b + 4), nullptr, nullptr, nullptr, qkv, T, 1536, 0);
    packa(qkv, Apack, Mt, 512, 512, 1536, 1024);                  // v slice (softmax(len1)=1 => o=v)
    gemm(Apack, WoutWp[l], Cbuf, Mt, 32, 16);
    epi(Cbuf, P(b + 5), hbuf, P(b + 10), P(b + 2), h2, T, 512, 3);
    packa(h2, Apack, Mt, 512, 512, 512, 0);
    gemm(Apack, ff1Wp[l], Cbuf, Mt, 128, 16);
    epi(Cbuf, P(b + 7), nullptr, nullptr, nullptr, tb, T, 2048, 4);
    packa(tb, Apack, Mt, 2048, 2048, 2048, 0);
    gemm(Apack, ff2Wp[l], Cbuf, Mt, 32, 64);
    epi(Cbuf, P(b + 9), h2, P(b + 11), P(b + 3), hbuf, T, 512, 3);
  }
  seq_mean_kernel<<<2, 256, 0, stream>>>(hbuf, seqv, T);

  // --- fusion: [mol_feats || seq] [N,1024] -> f [N,512] (f lives in d_out)
  {
    size_t total = (size_t)Mn * 32 * 512;
    pack_a_combined_kernel<<<(unsigned)((total + 255) / 256), 256, 0, stream>>>(
        x, seqv, Apack, Mn);
  }
  gemm(Apack, fusWp, Cbuf, Mn, 32, 32);
  epi(Cbuf, P(16), nullptr, nullptr, nullptr, fout, N, 512, 0);

  // --- SafeResidualBlock: out = f + gelu(clip(LN(nan2num(f@W+b)),+-10))
  packa(fout, Apack, Mn, 512, 512, 512, 0);
  gemm(Apack, resWp, Cbuf, Mn, 32, 16);
  epi(Cbuf, P(52), fout, P(54), P(53), fout, N, 512, 5);
}
